// NeuronS3DiffUpsample2D_79620103733631
// MI455X (gfx1250) — matrix operand order repaired and checked by reading, not compiled, not run on hardware
//
#include <hip/hip_runtime.h>
#include <hip/hip_bf16.h>

// ---------------------------------------------------------------------------
// DeMod-LoRA 3x3 conv on 2x nearest-upsampled input (CDNA5 / gfx1250, wave32).
//
// Identity: 3x3 conv on 2x-nearest-upsample == 4 parity-specific 2x2 convs on
// the ORIGINAL 160x160 image with folded weights (2.25x fewer FLOPs).
//   out(2i+ry, 2j+rx):  src row sy = i + ky2 + ry - 1, col sx = j + kx2 + rx - 1
//   folded tap (ky2,kx2) = sum of taps (dy,dx) with
//     ky2 = ((ry+dy-1)>>1)+1-ry,  kx2 = ((rx+dx-1)>>1)+1-rx
// Per workgroup: ONE 3-row halo staged in LDS serves BOTH output rows 2i,2i+1
// (ry=0 needs rows {i-1,i}, ry=1 needs {i,i+1}).  128 WMMAs/wave per barrier.
// ---------------------------------------------------------------------------

#define BB    8
#define CC    128
#define HH    160
#define WW    160
#define H2    320
#define W2    320
#define RANK  32
#define KTOT  1152            // unfolded K (f32 merged weights)
#define KF    2048            // folded K per (b,o): 4 parities * 4 taps * 128
#define HW    (HH*WW)         // 25600

// LDS halo: 3 rows x 34 cols x 128 ch bf16; pixel stride padded to 136 ushorts
// (272 B = 68 dwords) so 16-lane fragment reads spread across the 64 banks.
#define PIXSTRIDE 136
#define NPIX      (3*34)      // 102
#define NUNITS    (NPIX*16)   // 16B units to stage: 1632

typedef __attribute__((ext_vector_type(16))) __bf16 v16bf;
typedef __attribute__((ext_vector_type(8)))  __bf16 v8bf;
typedef __attribute__((ext_vector_type(8)))  float  v8f;

static __device__ __forceinline__ unsigned short f2bf(float f) {
    unsigned int u = __float_as_uint(f);
    unsigned int r = u + 0x7fffu + ((u >> 16) & 1u);   // round-to-nearest-even
    return (unsigned short)(r >> 16);
}

// ---------------------------------------------------------------------------
// Kernel X: x [B,C,H,W] f32 -> xT [B,H,W,C] bf16. grid (HW/64, C/32, B), 256
// ---------------------------------------------------------------------------
__global__ __launch_bounds__(256) void k_transpose_nhwc_bf16(
    const float* __restrict__ x, unsigned short* __restrict__ xT)
{
    __shared__ float tile[32][65];
    const int b  = blockIdx.z;
    const int c0 = blockIdx.y * 32;
    const int p0 = blockIdx.x * 64;
    const int t  = threadIdx.x;

    const int pp = t & 63;
    const int cb = t >> 6;
    #pragma unroll
    for (int cc = cb; cc < 32; cc += 4)
        tile[cc][pp] = x[((size_t)(b * CC + c0 + cc)) * HW + p0 + pp];
    __syncthreads();

    const int pp2 = t >> 2;
    const int ci  = (t & 3) * 8;
    unsigned short tmp[8];
    #pragma unroll
    for (int k = 0; k < 8; ++k) tmp[k] = f2bf(tile[ci + k][pp2]);
    uint4 v;
    v.x = (unsigned)tmp[0] | ((unsigned)tmp[1] << 16);
    v.y = (unsigned)tmp[2] | ((unsigned)tmp[3] << 16);
    v.z = (unsigned)tmp[4] | ((unsigned)tmp[5] << 16);
    v.w = (unsigned)tmp[6] | ((unsigned)tmp[7] << 16);
    unsigned short* dst = xT + ((size_t)b * HW + p0 + pp2) * CC + c0 + ci;
    *(uint4*)dst = v;
}

// ---------------------------------------------------------------------------
// Kernel W1: wm[o][dy3dx*128+i] = Wb + 0.25*(lora_up @ lora_down). grid 128x128
// ---------------------------------------------------------------------------
__global__ __launch_bounds__(128) void k_lora_merge(
    const float* __restrict__ Wb, const float* __restrict__ lora_up,
    const float* __restrict__ lora_down, float* __restrict__ wm)
{
    const int o = blockIdx.x;
    const int i = threadIdx.x;
    float acc[9];
    #pragma unroll
    for (int k9 = 0; k9 < 9; ++k9) acc[k9] = Wb[((size_t)(o * CC + i)) * 9 + k9];
    for (int r = 0; r < RANK; ++r) {
        const float su = 0.25f * lora_up[o * RANK + r];
        const float* ld = lora_down + ((size_t)(r * CC + i)) * 9;
        #pragma unroll
        for (int k9 = 0; k9 < 9; ++k9) acc[k9] += su * ld[k9];
    }
    #pragma unroll
    for (int k9 = 0; k9 < 9; ++k9) wm[(size_t)o * KTOT + k9 * CC + i] = acc[k9];
}

// ---------------------------------------------------------------------------
// Kernel W2: per (b,o): modulate by de_mod, demod = rsqrt(sum sq of UNFOLDED),
// fold 3x3 taps into 4 parity-specific 2x2 kernels, write bf16.
// wfold layout: [b][o][py*2+px][tap=ky2*2+kx2][i].  grid (128, 8), block 128
// ---------------------------------------------------------------------------
__global__ __launch_bounds__(128) void k_demod_fold_bf16(
    const float* __restrict__ wm, const float* __restrict__ de_mod,
    unsigned short* __restrict__ wfold)
{
    const int o = blockIdx.x;
    const int b = blockIdx.y;
    const int i = threadIdx.x;

    const float s = de_mod[b * CC + i];
    float v[9];
    float sum = 0.f;
    #pragma unroll
    for (int k9 = 0; k9 < 9; ++k9) {
        float t = wm[(size_t)o * KTOT + k9 * CC + i] * s;
        v[k9] = t;
        sum += t * t;
    }
    __shared__ float red[128];
    red[i] = sum;
    __syncthreads();
    #pragma unroll
    for (int st = 64; st > 0; st >>= 1) {
        if (i < st) red[i] += red[i + st];
        __syncthreads();
    }
    const float dm = rsqrtf(red[0] + 1e-8f);

    unsigned short* wrow = wfold + ((size_t)(b * CC + o)) * KF;
    #pragma unroll
    for (int py = 0; py < 2; ++py) {
        #pragma unroll
        for (int px = 0; px < 2; ++px) {
            float f[2][2] = {{0.f, 0.f}, {0.f, 0.f}};
            #pragma unroll
            for (int dy = 0; dy < 3; ++dy) {
                const int ky2 = ((py + dy - 1) >> 1) + 1 - py;   // 0 or 1
                #pragma unroll
                for (int dx = 0; dx < 3; ++dx) {
                    const int kx2 = ((px + dx - 1) >> 1) + 1 - px;
                    f[ky2][kx2] += v[dy * 3 + dx];
                }
            }
            #pragma unroll
            for (int tap = 0; tap < 4; ++tap)
                wrow[(py * 2 + px) * 512 + tap * CC + i] =
                    f2bf(f[tap >> 1][tap & 1] * dm);
        }
    }
}

// ---------------------------------------------------------------------------
// Main kernel. Per workgroup: batch b, source row i (output rows 2i, 2i+1),
// 32 source cols j0..j0+31 -> 2 rows x 64 output pixels, all 128 out channels.
// 8 waves, M-tile 16 each, 8 accumulators [ry][rx][nt].
// Stage 3x34x128 bf16 halo once, one barrier, then 128 barrier-free WMMAs.
// grid (5, 160, 8), block 256
// ---------------------------------------------------------------------------
__global__ __launch_bounds__(256) void k_conv_wmma(
    const unsigned short* __restrict__ xT,     // [B,H,W,C] bf16 bits
    const unsigned short* __restrict__ wfold,  // [B,O,KF] bf16 bits
    const float* __restrict__ bias,
    float* __restrict__ out)                   // [B,O,320,320] f32
{
    __shared__ __align__(32) unsigned short ldsT[NPIX * PIXSTRIDE]; // 27744 B

    const int b  = blockIdx.z;
    const int iy = blockIdx.y;          // source row
    const int j0 = blockIdx.x * 32;

    const int t     = threadIdx.x;
    const int lane  = t & 31;
    const int wv    = t >> 5;
    const int Mtile = wv * 16;
    const bool hi   = lane >= 16;
    const int  lm   = lane & 15;

    // ---- stage halo: rows iy-1..iy+1, cols j0-1..j0+32, all 128 channels ---
    for (int u = t; u < NUNITS; u += 256) {
        const int pix  = u >> 4;          // 0..101
        const int part = u & 15;          // 16B unit within 256B pixel
        const int row  = pix / 34;
        const int col  = pix - row * 34;
        const int sy   = iy - 1 + row;
        const int sx   = j0 - 1 + col;
        uint4 val = make_uint4(0u, 0u, 0u, 0u);
        if ((unsigned)sy < (unsigned)HH && (unsigned)sx < (unsigned)WW)
            val = *(const uint4*)(xT + (((size_t)b * HH + sy) * WW + sx) * CC
                                  + part * 8);
        *(uint4*)(ldsT + (size_t)pix * PIXSTRIDE + part * 8) = val;
    }
    __syncthreads();

    // ---- barrier-free WMMA loop: 2 ry x 2 rx x 4 taps x 4 K-chunks x 2 nt --
    const unsigned short* wrow =
        wfold + ((size_t)(b * CC + Mtile + lm)) * KF + (hi ? 8 : 0);

    v8f acc[8] = {};   // [ry*4 + rx*2 + nt]

    #pragma unroll
    for (int ry = 0; ry < 2; ++ry) {
        #pragma unroll
        for (int rx = 0; rx < 2; ++rx) {
            #pragma unroll
            for (int tap = 0; tap < 4; ++tap) {
                const int ky2  = tap >> 1;
                const int kx2  = tap & 1;
                const int lrow = ky2 + ry;           // 0..2 staged row
                const int cof  = kx2 + rx;           // col offset 0..2
                #pragma unroll
                for (int c = 0; c < 4; ++c) {
                    const int i0 = c * 32;
                    // A frag: lane holds M=lm, K={0..7,16..23}/{8..15,24..31}
                    const unsigned short* ap =
                        wrow + (ry * 2 + rx) * 512 + tap * CC + i0;
                    v8bf alo = *(const v8bf*)(ap);
                    v8bf ahi = *(const v8bf*)(ap + 16);
                    v16bf afrag = __builtin_shufflevector(
                        alo, ahi, 0,1,2,3,4,5,6,7,8,9,10,11,12,13,14,15);
                    // B frags: lane holds N=lm, K={0..7,16..23}/{8..15,24..31}
                    #pragma unroll
                    for (int nt = 0; nt < 2; ++nt) {
                        const int n = nt * 16 + lm;
                        const unsigned short* lp =
                            ldsT + (size_t)((lrow * 34) + n + cof) * PIXSTRIDE
                                 + i0 + (hi ? 8 : 0);
                        v8bf blo = *(const v8bf*)(lp);
                        v8bf bhi = *(const v8bf*)(lp + 16);
                        v16bf bfrag = __builtin_shufflevector(
                            blo, bhi, 0,1,2,3,4,5,6,7,8,9,10,11,12,13,14,15);
                        const int a = ry * 4 + rx * 2 + nt;
                        acc[a] = __builtin_amdgcn_wmma_f32_16x16x32_bf16(
                            false, afrag, false, bfrag, (short)0,
                            acc[a], false, false);
                    }
                }
            }
        }
    }

    // ---- epilogue: pair rx=0/rx=1 into float2, write both output rows ------
    // C layout: element v -> M = Mtile + v + (hi?8:0), N = lm
    #pragma unroll
    for (int ry = 0; ry < 2; ++ry) {
        const int yo = 2 * iy + ry;
        #pragma unroll
        for (int nt = 0; nt < 2; ++nt) {
            const int xc = 2 * (j0 + nt * 16 + lm);   // even output col
            #pragma unroll
            for (int v = 0; v < 8; ++v) {
                const int o = Mtile + v + (hi ? 8 : 0);
                const float bv = bias[o];
                float2 val;
                val.x = acc[ry * 4 + nt][v] + bv;       // rx = 0
                val.y = acc[ry * 4 + 2 + nt][v] + bv;   // rx = 1
                *(float2*)(out + (((size_t)(b * CC + o)) * H2 + yo) * W2 + xc)
                    = val;
            }
        }
    }
}

// ---------------------------------------------------------------------------
// Host launcher. Inputs: 0:x 1:de_mod 2:Wb 3:lora_up 4:lora_down 5:bias
// Output: [8,128,320,320] f32
// ---------------------------------------------------------------------------
extern "C" void kernel_launch(void* const* d_in, const int* in_sizes, int n_in,
                              void* d_out, int out_size, void* d_ws, size_t ws_size,
                              hipStream_t stream) {
    (void)in_sizes; (void)n_in; (void)out_size; (void)ws_size;
    const float* x         = (const float*)d_in[0];
    const float* de_mod    = (const float*)d_in[1];
    const float* Wb        = (const float*)d_in[2];
    const float* lora_up   = (const float*)d_in[3];
    const float* lora_down = (const float*)d_in[4];
    const float* bias      = (const float*)d_in[5];
    float* out = (float*)d_out;

    char* ws = (char*)d_ws;
    unsigned short* xT    = (unsigned short*)(ws);             // 52,428,800 B
    float*          wm    = (float*)(ws + 52428800);           //    589,824 B
    unsigned short* wfold = (unsigned short*)(ws + 53018624);  //  4,194,304 B

    k_transpose_nhwc_bf16<<<dim3(HW / 64, CC / 32, BB), 256, 0, stream>>>(x, xT);
    k_lora_merge<<<dim3(CC), 128, 0, stream>>>(Wb, lora_up, lora_down, wm);
    k_demod_fold_bf16<<<dim3(CC, BB), 128, 0, stream>>>(wm, de_mod, wfold);
    k_conv_wmma<<<dim3(WW / 32, HH, BB), 256, 0, stream>>>(xT, wfold, bias, out);
}
